// GruBlock_39573828666303
// MI455X (gfx1250) — matrix operand order repaired and checked by reading, not compiled, not run on hardware
//
#include <hip/hip_runtime.h>
#include <math.h>

// ---------------------------------------------------------------------------
// GRU block for MI455X (gfx1250, wave32, WMMA).
// Heavy math = two 25.8-GFLOP GEMMs done with v_wmma_f32_16x16x32_bf16:
//   gi  = g @ w_ih^T           (M=1024, N=6144, K=2048), parallel
//   gh  = h @ w_hh^T per step  (M=16,   N=6144, K=2048), 64 sequential steps
// Weights converted once fp32->bf16 (25 MB each) so the recurrent GEMM streams
// w_hh from the 192 MB L2 every step instead of HBM. fp32 accumulation.
// Workspace requirement: ~88.8 MB.
// ---------------------------------------------------------------------------

typedef __attribute__((ext_vector_type(16))) __bf16 v16bf;
typedef __attribute__((ext_vector_type(8)))  __bf16 v8bf;
typedef __attribute__((ext_vector_type(8)))  float  v8f;

#define B_    16
#define S_    64
#define CIN_  64
#define HID_  128
#define GC_   32
#define G_    2048
#define G3_   6144
#define PX_   64        // H*W = 8*8
#define BS_   (B_*S_)   // 1024

__device__ __forceinline__ float celu1(float v) {
    return v > 0.0f ? v : expm1f(v);
}

// ------------------------- fp32 -> bf16 convert ----------------------------
__global__ void f32_to_bf16_kernel(const float* __restrict__ src,
                                   __bf16* __restrict__ dst, int n) {
    int i = blockIdx.x * 256 + threadIdx.x;
    if (i < n) dst[i] = (__bf16)src[i];
}

// ---------------- fused front conv stack (per image, LDS-resident) ---------
// x[img,64,8,8] -> cin(1x1)+celu -> dw3x3+celu (x2) -> gruin(1x1)+celu -> g bf16
__global__ void conv_front_kernel(const float* __restrict__ x,
                                  const float* __restrict__ cin_w,
                                  const float* __restrict__ cin_b,
                                  const float* __restrict__ dw_w,
                                  const float* __restrict__ dw_b,
                                  const float* __restrict__ gruin_w,
                                  const float* __restrict__ gruin_b,
                                  __bf16* __restrict__ g_bf) {
    extern __shared__ float sm[];
    float* xs = sm;            // 64*64
    float* ha = sm + 4096;     // 128*64
    float* hb = sm + 12288;    // 128*64
    const int img = blockIdx.x;
    const int tid = threadIdx.x;

    const float* xi = x + (size_t)img * CIN_ * PX_;
    for (int t = tid; t < CIN_ * PX_; t += 256) xs[t] = xi[t];
    __syncthreads();

    // cin 1x1 (64->128) + celu
    for (int t = tid; t < HID_ * PX_; t += 256) {
        int o = t >> 6, px = t & 63;
        float acc = cin_b[o];
        const float* wrow = cin_w + o * CIN_;
        #pragma unroll 8
        for (int c = 0; c < CIN_; ++c) acc += wrow[c] * xs[c * PX_ + px];
        ha[t] = celu1(acc);
    }
    __syncthreads();

    // 2 depthwise 3x3 layers + celu (ping-pong ha<->hb)
    float* src = ha; float* dst = hb;
    for (int l = 0; l < 2; ++l) {
        for (int t = tid; t < HID_ * PX_; t += 256) {
            int ch = t >> 6, py = (t >> 3) & 7, pxx = t & 7;
            float acc = dw_b[l * HID_ + ch];
            const float* w = dw_w + ((size_t)l * HID_ + ch) * 9;
            #pragma unroll
            for (int dy = 0; dy < 3; ++dy)
                #pragma unroll
                for (int dx = 0; dx < 3; ++dx) {
                    int yy = py + dy - 1, xx = pxx + dx - 1;
                    if (yy >= 0 && yy < 8 && xx >= 0 && xx < 8)
                        acc += w[dy * 3 + dx] * src[ch * PX_ + yy * 8 + xx];
                }
            dst[t] = celu1(acc);
        }
        __syncthreads();
        float* tp = src; src = dst; dst = tp;
    }

    // gruin 1x1 (128->32) + celu -> g[b,s, cgc*64+px] as bf16
    for (int t = tid; t < GC_ * PX_; t += 256) {
        int o = t >> 6, px = t & 63;
        float acc = gruin_b[o];
        const float* wrow = gruin_w + o * HID_;
        #pragma unroll 8
        for (int c = 0; c < HID_; ++c) acc += wrow[c] * src[c * PX_ + px];
        g_bf[(size_t)img * G_ + t] = (__bf16)celu1(acc);
    }
}

// ------------------- gi GEMM: gi[s,b,:] = g[b,s,:] @ w_ih^T + b_ih ---------
// grid (96, 64), block 128 (4 waves); each wave computes one 16x16 tile.
__global__ void gemm_gi_kernel(const __bf16* __restrict__ gbf,
                               const __bf16* __restrict__ wih,
                               const float* __restrict__ b_ih,
                               float* __restrict__ gi) {
    const int lane  = threadIdx.x & 31;
    const int ntile = blockIdx.x * 4 + (threadIdx.x >> 5);  // 0..383
    const int mtile = blockIdx.y;                           // 0..63
    const int m = lane & 15, half = lane >> 4;

    const __bf16* Arow = gbf + (size_t)(mtile * 16 + m) * G_;
    const __bf16* Brow = wih + (size_t)(ntile * 16 + m) * G_;  // row n of w_ih

    v8f c = {};
    for (int k0 = 0; k0 < G_; k0 += 32) {
        union { v16bf v; v8bf h[2]; } a;
        a.h[0] = *(const v8bf*)(Arow + k0 + 8 * half);        // K = 8*half + 0..7
        a.h[1] = *(const v8bf*)(Arow + k0 + 16 + 8 * half);   // K = 16+8*half + 0..7
        union { v16bf v; v8bf h[2]; } b;
        b.h[0] = *(const v8bf*)(Brow + k0 + 8 * half);        // K = 8*half + 0..7
        b.h[1] = *(const v8bf*)(Brow + k0 + 16 + 8 * half);   // K = 16+8*half + 0..7
        c = __builtin_amdgcn_wmma_f32_16x16x32_bf16(false, a.v, false, b.v,
                                                    (short)0, c, false, false);
    }
    const int col = ntile * 16 + m;   // lane's N index == m
    const float bias = b_ih[col];
    #pragma unroll
    for (int i = 0; i < 8; ++i) {
        int row = mtile * 16 + i + 8 * half;   // row = b*64 + s
        int bb = row >> 6, ss = row & 63;
        gi[(size_t)(ss * 16 + bb) * G3_ + col] = c[i] + bias;
    }
}

// ------------ recurrent step GEMM: gh[16,6144] = hmask @ w_hh^T ------------
// grid 96, block 128 (4 waves); one 16x16 tile per wave, K=2048.
__global__ void gemm_step_kernel(const __bf16* __restrict__ hbf,
                                 const __bf16* __restrict__ whh,
                                 float* __restrict__ gh) {
    const int lane  = threadIdx.x & 31;
    const int ntile = blockIdx.x * 4 + (threadIdx.x >> 5);  // 0..383
    const int m = lane & 15, half = lane >> 4;

    const __bf16* Arow = hbf + (size_t)m * G_;
    const __bf16* Brow = whh + (size_t)(ntile * 16 + m) * G_;

    v8f c = {};
    for (int k0 = 0; k0 < G_; k0 += 32) {
        union { v16bf v; v8bf h[2]; } a;
        a.h[0] = *(const v8bf*)(Arow + k0 + 8 * half);
        a.h[1] = *(const v8bf*)(Arow + k0 + 16 + 8 * half);
        union { v16bf v; v8bf h[2]; } b;
        b.h[0] = *(const v8bf*)(Brow + k0 + 8 * half);
        b.h[1] = *(const v8bf*)(Brow + k0 + 16 + 8 * half);
        c = __builtin_amdgcn_wmma_f32_16x16x32_bf16(false, a.v, false, b.v,
                                                    (short)0, c, false, false);
    }
    const int col = ntile * 16 + m;
    #pragma unroll
    for (int i = 0; i < 8; ++i)
        gh[(size_t)(i + 8 * half) * G3_ + col] = c[i];
}

// ------------------------------- small kernels -----------------------------
__global__ void init_h_kernel(const float* __restrict__ state,
                              float* __restrict__ hbuf) {
    int t = blockIdx.x * 256 + threadIdx.x;
    if (t < B_ * G_) hbuf[t] = state[t];
}

__global__ void mask_h_kernel(const float* __restrict__ hbuf,
                              const float* __restrict__ mask, int s,
                              float* __restrict__ hm, __bf16* __restrict__ hmbf) {
    int t = blockIdx.x * 256 + threadIdx.x;
    if (t >= B_ * G_) return;
    int b = t >> 11;
    float v = hbuf[t] * mask[b * S_ + s];
    hm[t]   = v;
    hmbf[t] = (__bf16)v;
}

__global__ void gru_gates_kernel(const float* __restrict__ gi,
                                 const float* __restrict__ gh,
                                 const float* __restrict__ b_hh,
                                 const float* __restrict__ hmask, int s,
                                 float* __restrict__ hbuf,
                                 float* __restrict__ gseq) {
    int t = blockIdx.x * 256 + threadIdx.x;
    if (t >= B_ * G_) return;
    int b = t >> 11, g = t & (G_ - 1);
    const float* gis = gi + (size_t)(s * 16 + b) * G3_;
    float ir = gis[g], iz = gis[G_ + g], in_ = gis[2 * G_ + g];
    const float* ghb = gh + (size_t)b * G3_;
    float hr = ghb[g]          + b_hh[g];
    float hz = ghb[G_ + g]     + b_hh[G_ + g];
    float hn = ghb[2 * G_ + g] + b_hh[2 * G_ + g];
    float hx = hmask[t];
    float r = 1.0f / (1.0f + expf(-(ir + hr)));
    float z = 1.0f / (1.0f + expf(-(iz + hz)));
    float n = tanhf(in_ + r * hn);
    float hnew = (1.0f - z) * n + z * hx;
    hbuf[t] = hnew;
    gseq[(size_t)(b * S_ + s) * G_ + g] = hnew;   // [B,S,G]
}

__global__ void copy_hg_kernel(const float* __restrict__ hbuf,
                               float* __restrict__ hg) {
    int t = blockIdx.x * 256 + threadIdx.x;
    if (t < B_ * G_) hg[t] = hbuf[t];
}

// ---------------- fused back end: gruout 1x1 -> cout 1x1 -> +x -> celu -----
__global__ void out_back_kernel(const float* __restrict__ gseq,
                                const float* __restrict__ x,
                                const float* __restrict__ go_w,
                                const float* __restrict__ go_b,
                                const float* __restrict__ co_w,
                                const float* __restrict__ co_b,
                                float* __restrict__ y) {
    __shared__ float gs[GC_ * PX_];    // 8 KB
    __shared__ float h2[HID_ * PX_];   // 32 KB
    const int img = blockIdx.x, tid = threadIdx.x;

    for (int t = tid; t < GC_ * PX_; t += 256)
        gs[t] = gseq[(size_t)img * G_ + t];
    __syncthreads();

    for (int t = tid; t < HID_ * PX_; t += 256) {
        int o = t >> 6, px = t & 63;
        float acc = go_b[o];
        const float* wrow = go_w + o * GC_;
        #pragma unroll 8
        for (int c = 0; c < GC_; ++c) acc += wrow[c] * gs[c * PX_ + px];
        h2[t] = acc;
    }
    __syncthreads();

    for (int t = tid; t < CIN_ * PX_; t += 256) {
        int o = t >> 6, px = t & 63;
        float acc = co_b[o];
        const float* wrow = co_w + o * HID_;
        #pragma unroll 8
        for (int c = 0; c < HID_; ++c) acc += wrow[c] * h2[c * PX_ + px];
        acc += x[(size_t)img * CIN_ * PX_ + t];
        y[(size_t)img * CIN_ * PX_ + t] = celu1(acc);
    }
}

// ---------------------------------------------------------------------------
extern "C" void kernel_launch(void* const* d_in, const int* in_sizes, int n_in,
                              void* d_out, int out_size, void* d_ws, size_t ws_size,
                              hipStream_t stream) {
    (void)in_sizes; (void)n_in; (void)out_size; (void)ws_size;

    const float* x        = (const float*)d_in[0];
    const float* state    = (const float*)d_in[1];
    const float* mask     = (const float*)d_in[2];
    const float* cin_w    = (const float*)d_in[3];
    const float* cin_b    = (const float*)d_in[4];
    const float* dw_w     = (const float*)d_in[5];
    const float* dw_b     = (const float*)d_in[6];
    const float* gruin_w  = (const float*)d_in[7];
    const float* gruin_b  = (const float*)d_in[8];
    const float* w_ih     = (const float*)d_in[9];
    const float* w_hh     = (const float*)d_in[10];
    const float* b_ih     = (const float*)d_in[11];
    const float* b_hh     = (const float*)d_in[12];
    const float* gruout_w = (const float*)d_in[13];
    const float* gruout_b = (const float*)d_in[14];
    const float* cout_w   = (const float*)d_in[15];
    const float* cout_b   = (const float*)d_in[16];
    float* out = (float*)d_out;

    // workspace layout (bytes); total ~88.8 MB
    char* ws = (char*)d_ws;
    const size_t W_ELEMS = (size_t)G3_ * G_;           // 12,582,912
    __bf16* wih_bf  = (__bf16*)(ws);
    __bf16* whh_bf  = (__bf16*)(ws + W_ELEMS * 2);
    __bf16* g_bf    = (__bf16*)(ws + W_ELEMS * 4);
    float*  gi      = (float*)(ws + W_ELEMS * 4 + (size_t)BS_ * G_ * 2);
    float*  gseq    = (float*)((char*)gi   + (size_t)S_ * B_ * G3_ * 4);
    float*  gh      = (float*)((char*)gseq + (size_t)B_ * S_ * G_ * 4);
    float*  hbuf    = (float*)((char*)gh   + (size_t)B_ * G3_ * 4);
    float*  hmask   = (float*)((char*)hbuf + (size_t)B_ * G_ * 4);
    __bf16* hmbf    = (__bf16*)((char*)hmask + (size_t)B_ * G_ * 4);

    // 1) convert GRU weights to bf16 (once per call; weights fit in L2 in bf16)
    {
        int n = (int)W_ELEMS;
        int blocks = (n + 255) / 256;
        f32_to_bf16_kernel<<<blocks, 256, 0, stream>>>(w_ih, wih_bf, n);
        f32_to_bf16_kernel<<<blocks, 256, 0, stream>>>(w_hh, whh_bf, n);
    }

    // 2) fused front conv stack -> g (bf16), one block per image
    conv_front_kernel<<<BS_, 256, 81920, stream>>>(
        x, cin_w, cin_b, dw_w, dw_b, gruin_w, gruin_b, g_bf);

    // 3) gi = g @ w_ih^T + b_ih  (bf16 WMMA, fp32 accum)
    gemm_gi_kernel<<<dim3(96, 64), 128, 0, stream>>>(g_bf, wih_bf, b_ih, gi);

    // 4) recurrent scan, 64 steps: mask -> WMMA GEMM -> gates
    init_h_kernel<<<(B_ * G_ + 255) / 256, 256, 0, stream>>>(state, hbuf);
    for (int s = 0; s < S_; ++s) {
        mask_h_kernel<<<(B_ * G_ + 255) / 256, 256, 0, stream>>>(
            hbuf, mask, s, hmask, hmbf);
        gemm_step_kernel<<<96, 128, 0, stream>>>(hmbf, whh_bf, gh);
        gru_gates_kernel<<<(B_ * G_ + 255) / 256, 256, 0, stream>>>(
            gi, gh, b_hh, hmask, s, hbuf, gseq);
    }
    copy_hg_kernel<<<(B_ * G_ + 255) / 256, 256, 0, stream>>>(
        hbuf, out + (size_t)BS_ * CIN_ * PX_);

    // 5) fused back end: gruout -> cout -> +x -> celu
    out_back_kernel<<<BS_, 256, 0, stream>>>(
        gseq, x, gruout_w, gruout_b, cout_w, cout_b, out);
}
